// FullyConnectedSE3_55851754717696
// MI455X (gfx1250) — matrix operand order repaired and checked by reading, not compiled, not run on hardware
//
#include <hip/hip_runtime.h>
#include <hip/hip_bf16.h>
#include <cstdint>

// ---------------------------------------------------------------------------
// FullyConnectedSE3 for MI455X (gfx1250, wave32, WMMA + TDM).
// GEMM stages: V_WMMA_F32_16X16X32_F16 (f16 operands, f32 accumulate).
// Edge tensor materialized once as f16 (32MB -> L2-resident); attention
// brings each 512x64 row-block into LDS via tensor_load_to_lds (TENSORcnt).
// ---------------------------------------------------------------------------

typedef _Float16 f16;
typedef __attribute__((ext_vector_type(16))) _Float16 v16h;
typedef __attribute__((ext_vector_type(8)))  _Float16 v8h;
typedef __attribute__((ext_vector_type(8)))  float    v8f;
typedef __attribute__((ext_vector_type(4)))  unsigned int u32x4;
typedef __attribute__((ext_vector_type(8)))  int      i32x8;
typedef __attribute__((ext_vector_type(4)))  int      i32x4;

#define LSEQ 512

#if defined(__has_builtin)
#if __has_builtin(__builtin_amdgcn_tensor_load_to_lds) && \
    __has_builtin(__builtin_amdgcn_s_wait_tensorcnt)
#define HAVE_TDM 1
#endif
#endif

__device__ __forceinline__ v8f wmma16(v16h a, v16h b, v8f c) {
  // D = A(16x32 f16) * B(32x16 f16) + C(16x16 f32)
  return __builtin_amdgcn_wmma_f32_16x16x32_f16(false, a, false, b, (short)0, c,
                                                false, false);
}

// ISA 16-bit A-matrix 16x32 lane layout: element slot for (m, k) within chunk.
__device__ __forceinline__ int a_slot(int m, int k) {
  int lane = m + (((k >> 3) & 1) << 4);
  int e = ((k >> 4) << 3) + (k & 7);
  return lane * 16 + e;
}
// ISA 16-bit B-matrix 32x16 lane layout: element slot for (k, n) within chunk.
__device__ __forceinline__ int b_slot(int k, int n) {
  int lane = n + (((k >> 3) & 1) << 4);
  return lane * 16 + ((k >> 4) << 3) + (k & 7);
}

// A fragment straight from a row-major f16 row: per lane two contiguous
// 16B runs (K = koff..koff+7 and koff+16..koff+23), koff = 32*chunk + 8*hi.
__device__ __forceinline__ v16h load_a_frag(const f16* row, int koff) {
  v8h p0 = *(const v8h*)(row + koff);
  v8h p1 = *(const v8h*)(row + koff + 16);
  return __builtin_shufflevector(p0, p1, 0, 1, 2, 3, 4, 5, 6, 7, 8, 9, 10, 11,
                                 12, 13, 14, 15);
}

union HPack {
  f16 h[8];
  u32x4 u;
};

__device__ __forceinline__ float rsum8(float v) {
  v += __shfl_xor(v, 1, 8);
  v += __shfl_xor(v, 2, 8);
  v += __shfl_xor(v, 4, 8);
  return v;
}
__device__ __forceinline__ float wsum32(float v) {
  for (int m = 16; m; m >>= 1) v += __shfl_xor(v, m, 32);
  return v;
}
__device__ __forceinline__ float wmax32(float v) {
  for (int m = 16; m; m >>= 1) v = fmaxf(v, __shfl_xor(v, m, 32));
  return v;
}

__device__ __forceinline__ float red64(float v, float* buf) {
  int t = threadIdx.x;
  buf[t] = v;
  __syncthreads();
  if (t < 32) {
    float x = buf[t] + buf[t + 32];
    x = wsum32(x);
    if (t == 0) buf[0] = x;
  }
  __syncthreads();
  float r = buf[0];
  __syncthreads();
  return r;
}

// ---------------------------------------------------------------------------
// K1: node path (512 rows, ~38 MFLOP) -> scalar fp32.
// ---------------------------------------------------------------------------
__global__ __launch_bounds__(64) void node_kernel(
    const float* __restrict__ msa, const float* __restrict__ state,
    const float* __restrict__ nmg, const float* __restrict__ nmb,
    const float* __restrict__ nsg, const float* __restrict__ nsb,
    const float* __restrict__ embw, const float* __restrict__ embb,
    const float* __restrict__ fng, const float* __restrict__ fnb,
    const float* __restrict__ w1, const float* __restrict__ b1,
    const float* __restrict__ w2, const float* __restrict__ b2,
    const float* __restrict__ nng, const float* __restrict__ nnb,
    float* __restrict__ node) {
  __shared__ float sin_[320];
  __shared__ float buf[64];
  __shared__ float h1[128];
  __shared__ float nd[64];
  __shared__ float ln_[64];
  const int l = blockIdx.x, tid = threadIdx.x;

  float p0 = 0.f, p1 = 0.f;
  for (int k = tid; k < 256; k += 64) {
    float v = msa[l * 256 + k];
    p0 += v; p1 += v * v;
  }
  float s = red64(p0, buf), ss = red64(p1, buf);
  float mean = s * (1.f / 256.f);
  float rs = rsqrtf(ss * (1.f / 256.f) - mean * mean + 1e-5f);
  for (int k = tid; k < 256; k += 64)
    sin_[k] = (msa[l * 256 + k] - mean) * rs * nmg[k] + nmb[k];

  float v0 = state[l * 64 + tid];
  s = red64(v0, buf); ss = red64(v0 * v0, buf);
  mean = s * (1.f / 64.f);
  rs = rsqrtf(ss * (1.f / 64.f) - mean * mean + 1e-5f);
  sin_[256 + tid] = (v0 - mean) * rs * nsg[tid] + nsb[tid];
  __syncthreads();

  {
    float acc = embb[tid];
    for (int k = 0; k < 320; ++k) acc += sin_[k] * embw[k * 64 + tid];
    nd[tid] = acc;
  }
  __syncthreads();

  float x = nd[tid];
  s = red64(x, buf); ss = red64(x * x, buf);
  mean = s * (1.f / 64.f);
  rs = rsqrtf(ss * (1.f / 64.f) - mean * mean + 1e-5f);
  ln_[tid] = (x - mean) * rs * fng[tid] + fnb[tid];
  __syncthreads();
  for (int o = tid; o < 128; o += 64) {
    float acc = b1[o];
    for (int k = 0; k < 64; ++k) acc += ln_[k] * w1[k * 128 + o];
    h1[o] = fmaxf(acc, 0.f);
  }
  __syncthreads();
  {
    float acc = b2[tid];
    for (int k = 0; k < 128; ++k) acc += h1[k] * w2[k * 64 + tid];
    x = nd[tid] + acc;
  }
  s = red64(x, buf); ss = red64(x * x, buf);
  mean = s * (1.f / 64.f);
  rs = rsqrtf(ss * (1.f / 64.f) - mean * mean + 1e-5f);
  node[l * 64 + tid] = (x - mean) * rs * nng[tid] + nnb[tid];
}

// ---------------------------------------------------------------------------
// K2: edge pipeline. One block = 32 pair-rows (fixed i). WMMA chains for
// concat(193, pad 224) @ (224->64) and FFN 64->128(relu)->64.
// float4 global loads, 16B-packed LDS staging, 16B-packed f16 stores.
// ---------------------------------------------------------------------------
__global__ __launch_bounds__(256) void edge_kernel(
    const float* __restrict__ pair, const float* __restrict__ xyz,
    const int* __restrict__ idxp,
    const float* __restrict__ npg, const float* __restrict__ npb,
    const float* __restrict__ embw, const float* __restrict__ embb,
    const float* __restrict__ fng, const float* __restrict__ fnb,
    const float* __restrict__ w1, const float* __restrict__ b1,
    const float* __restrict__ w2, const float* __restrict__ b2,
    const float* __restrict__ neg_g, const float* __restrict__ neg_b,
    f16* __restrict__ edge_out) {
  extern __shared__ __align__(32) char smem[];
  char* sp = smem;
  auto carve = [&](size_t n) {
    char* p = sp;
    sp += (n + 31) & ~size_t(31);
    return p;
  };
  f16* bwE = (f16*)carve(7 * 4 * 512 * sizeof(f16));   // embed W, B layout
  f16* bw1 = (f16*)carve(2 * 8 * 512 * sizeof(f16));   // ffe W1
  f16* bw2 = (f16*)carve(4 * 4 * 512 * sizeof(f16));   // ffe W2
  f16* ast = (f16*)carve(7 * 2 * 512 * sizeof(f16));   // A staging [chunk][mt]
  float* pre = (float*)carve(32 * 64 * sizeof(float));
  float* s_npg = (float*)carve(128 * 4);
  float* s_npb = (float*)carve(128 * 4);
  float* s_eb = (float*)carve(64 * 4);
  float* s_b1 = (float*)carve(128 * 4);
  float* s_b2 = (float*)carve(64 * 4);
  float* s_fng = (float*)carve(64 * 4);
  float* s_fnb = (float*)carve(64 * 4);
  float* s_neg = (float*)carve(64 * 4);
  float* s_neb = (float*)carve(64 * 4);

  const int tid = threadIdx.x;
  const int wv = tid >> 5;
  const int lane = tid & 31;
  const int r0 = blockIdx.x * 32;
  const int i = r0 >> 9;
  const int j0 = r0 & 511;

  // Stage weights f32->f16 into ISA B layout.
  for (int t = tid; t < 224 * 64; t += 256) {
    int k = t >> 6, n = t & 63;
    float w = (k < 193) ? embw[k * 64 + n] : 0.f;
    bwE[(((k >> 5) << 2) + (n >> 4)) * 512 + b_slot(k & 31, n & 15)] = (f16)w;
  }
  for (int t = tid; t < 64 * 128; t += 256) {
    int k = t >> 7, n = t & 127;
    bw1[(((k >> 5) << 3) + (n >> 4)) * 512 + b_slot(k & 31, n & 15)] =
        (f16)w1[k * 128 + n];
  }
  for (int t = tid; t < 128 * 64; t += 256) {
    int k = t >> 6, n = t & 63;
    bw2[(((k >> 5) << 2) + (n >> 4)) * 512 + b_slot(k & 31, n & 15)] =
        (f16)w2[k * 64 + n];
  }
  for (int t = tid; t < 128; t += 256) {
    s_npg[t] = npg[t]; s_npb[t] = npb[t]; s_b1[t] = b1[t];
  }
  for (int t = tid; t < 64; t += 256) {
    s_eb[t] = embb[t]; s_b2[t] = b2[t]; s_fng[t] = fng[t]; s_fnb[t] = fnb[t];
    s_neg[t] = neg_g[t]; s_neb[t] = neg_b[t];
  }
  __syncthreads();

  // Build concat features (LN(pair) | rbf | neighbor | pad) as f16 A tiles.
  {
    const int rl = tid >> 3, t8 = tid & 7;
    const int j = j0 + rl;
    const size_t prow = ((size_t)i * 512 + j) * 128;
    const float4* pr = (const float4*)(pair + prow + t8 * 16);
    float4 q0 = pr[0], q1 = pr[1], q2 = pr[2], q3 = pr[3];
    float vals[16] = {q0.x, q0.y, q0.z, q0.w, q1.x, q1.y, q1.z, q1.w,
                      q2.x, q2.y, q2.z, q2.w, q3.x, q3.y, q3.z, q3.w};
    float p0 = 0.f, p1 = 0.f;
#pragma unroll
    for (int u = 0; u < 16; ++u) { p0 += vals[u]; p1 += vals[u] * vals[u]; }
    p0 = rsum8(p0); p1 = rsum8(p1);
    float mean = p0 * (1.f / 128.f);
    float rs = rsqrtf(p1 * (1.f / 128.f) - mean * mean + 1e-5f);
    const int mt = rl >> 4, mm = rl & 15;
#pragma unroll
    for (int g = 0; g < 2; ++g) {
      HPack pk;
      int kb_ = t8 * 16 + g * 8;
#pragma unroll
      for (int u = 0; u < 8; ++u) {
        int k = kb_ + u;
        pk.h[u] = (f16)((vals[g * 8 + u] - mean) * rs * s_npg[k] + s_npb[k]);
      }
      *(u32x4*)&ast[(((kb_ >> 5) << 1) + mt) * 512 + a_slot(mm, kb_ & 31)] =
          pk.u;
    }
    float cix = xyz[i * 9 + 3], ciy = xyz[i * 9 + 4], ciz = xyz[i * 9 + 5];
    float dx = xyz[j * 9 + 3] - cix, dy = xyz[j * 9 + 4] - ciy,
          dz = xyz[j * 9 + 5] - ciz;
    float D = sqrtf(dx * dx + dy * dy + dz * dz + 1e-8f);
    {
      HPack pk;
#pragma unroll
      for (int u = 0; u < 8; ++u) {
        int t = t8 * 8 + u;
        float mu = 2.f + (float)t * (20.f / 63.f);
        float z = (D - mu) * 3.2f;  // 1/sigma, sigma = 20/64
        pk.h[u] = (f16)__expf(-z * z);
      }
      int kb_ = 128 + t8 * 8;
      *(u32x4*)&ast[(((kb_ >> 5) << 1) + mt) * 512 + a_slot(mm, kb_ & 31)] =
          pk.u;
    }
    for (int k = 192 + t8; k < 224; k += 8) {
      float h = 0.f;
      if (k == 192) {
        int d = idxp[j] - idxp[i];
        float sgn = (d > 0) ? 1.f : ((d < 0) ? -1.f : 0.f);
        h = sgn * log1pf(fabsf((float)d)) * (1.f / 3.f);
      }
      ast[(((k >> 5) << 1) + mt) * 512 + a_slot(mm, k & 31)] = (f16)h;
    }
  }
  __syncthreads();

  // Embed matmul: 2 mtiles x 4 ntiles across 8 waves, K=224 (7 chunks).
  {
    const int mt = wv >> 2, nt = wv & 3;
    v8f acc = {};
#pragma unroll
    for (int c = 0; c < 7; ++c) {
      v16h a = *(const v16h*)(ast + ((c << 1) + mt) * 512 + lane * 16);
      v16h b = *(const v16h*)(bwE + ((c << 2) + nt) * 512 + lane * 16);
      acc = wmma16(a, b, acc);
    }
    const int n0 = nt * 16 + (lane & 15);
    const int mb = mt * 16 + ((lane >> 4) << 3);
#pragma unroll
    for (int r = 0; r < 8; ++r) pre[(mb + r) * 64 + n0] = acc[r] + s_eb[n0];
  }
  __syncthreads();

  // FFE pre-LN -> f16 A staging (chunks 0..1).
  {
    const int rl = tid >> 3, t8 = tid & 7;
    float vals[8];
    float p0 = 0.f, p1 = 0.f;
#pragma unroll
    for (int u = 0; u < 8; ++u) {
      float v = pre[rl * 64 + t8 * 8 + u];
      vals[u] = v; p0 += v; p1 += v * v;
    }
    p0 = rsum8(p0); p1 = rsum8(p1);
    float mean = p0 * (1.f / 64.f);
    float rs = rsqrtf(p1 * (1.f / 64.f) - mean * mean + 1e-5f);
    const int mt = rl >> 4, mm = rl & 15;
    HPack pk;
    int kb_ = t8 * 8;
#pragma unroll
    for (int u = 0; u < 8; ++u) {
      int k = kb_ + u;
      pk.h[u] = (f16)((vals[u] - mean) * rs * s_fng[k] + s_fnb[k]);
    }
    *(u32x4*)&ast[(((kb_ >> 5) << 1) + mt) * 512 + a_slot(mm, kb_ & 31)] = pk.u;
  }
  __syncthreads();

  // FFE stage 1: 64->128, relu, written as A tiles into chunks 2..5.
  for (int task = wv; task < 16; task += 8) {
    const int mt = task >> 3, nt = task & 7;
    v8f acc = {};
#pragma unroll
    for (int c = 0; c < 2; ++c) {
      v16h a = *(const v16h*)(ast + ((c << 1) + mt) * 512 + lane * 16);
      v16h b = *(const v16h*)(bw1 + ((c << 3) + nt) * 512 + lane * 16);
      acc = wmma16(a, b, acc);
    }
    const int n0 = nt * 16 + (lane & 15);
    const int mbase = (lane >> 4) << 3;
#pragma unroll
    for (int r = 0; r < 8; ++r) {
      float h = fmaxf(acc[r] + s_b1[n0], 0.f);
      int ch = 2 + (n0 >> 5);
      ast[((ch << 1) + mt) * 512 + a_slot(mbase + r, n0 & 31)] = (f16)h;
    }
  }
  __syncthreads();

  // FFE stage 2: 128->64, residual into pre.
  {
    const int mt = wv >> 2, nt = wv & 3;
    v8f acc = {};
#pragma unroll
    for (int c = 0; c < 4; ++c) {
      v16h a = *(const v16h*)(ast + (((2 + c) << 1) + mt) * 512 + lane * 16);
      v16h b = *(const v16h*)(bw2 + ((c << 2) + nt) * 512 + lane * 16);
      acc = wmma16(a, b, acc);
    }
    const int n0 = nt * 16 + (lane & 15);
    const int mb = mt * 16 + ((lane >> 4) << 3);
#pragma unroll
    for (int r = 0; r < 8; ++r) pre[(mb + r) * 64 + n0] += acc[r] + s_b2[n0];
  }
  __syncthreads();

  // Final edge LN -> packed f16 global store (16B per thread).
  {
    const int rl = tid >> 3, t8 = tid & 7;
    const int j = j0 + rl;
    float vals[8];
    float p0 = 0.f, p1 = 0.f;
#pragma unroll
    for (int u = 0; u < 8; ++u) {
      float v = pre[rl * 64 + t8 * 8 + u];
      vals[u] = v; p0 += v; p1 += v * v;
    }
    p0 = rsum8(p0); p1 = rsum8(p1);
    float mean = p0 * (1.f / 64.f);
    float rs = rsqrtf(p1 * (1.f / 64.f) - mean * mean + 1e-5f);
    HPack pk;
#pragma unroll
    for (int u = 0; u < 8; ++u) {
      int k = t8 * 8 + u;
      pk.h[u] = (f16)((vals[u] - mean) * rs * s_neg[k] + s_neb[k]);
    }
    f16* dst = edge_out + ((size_t)i * 512 + j) * 64;
    *(u32x4*)(dst + t8 * 8) = pk.u;
  }
}

// ---------------------------------------------------------------------------
// K3a: per-layer scalar prep: qb/kb/v0b/sself = s@W, vm/vself from v.
// ---------------------------------------------------------------------------
__global__ __launch_bounds__(128) void prep_kernel(
    const float* __restrict__ s_in, int C0, const float* __restrict__ v_in,
    int C1, const float* __restrict__ xyz, const float* __restrict__ wq,
    const float* __restrict__ wk, const float* __restrict__ wv0,
    const float* __restrict__ wself0, const float* __restrict__ wv1,
    const float* __restrict__ wself1, float* __restrict__ qb,
    float* __restrict__ kb, float* __restrict__ v0b, float* __restrict__ ssb,
    float* __restrict__ vm, float* __restrict__ vs) {
  __shared__ float srow[64];
  __shared__ float vrow[96];
  const int l = blockIdx.x, tid = threadIdx.x;
  if (tid < C0) srow[tid] = s_in[l * C0 + tid];
  if (v_in) {
    if (tid < C1 * 3) vrow[tid] = v_in[l * 96 + tid];
  } else if (tid < 9) {
    int c = tid / 3, x = tid % 3;
    vrow[c * 3 + x] = xyz[l * 9 + c * 3 + x] - xyz[l * 9 + 3 + x];
  }
  __syncthreads();
  {
    int mat = tid >> 5, d = tid & 31;
    const float* W =
        (mat == 0) ? wq : ((mat == 1) ? wk : ((mat == 2) ? wv0 : wself0));
    float acc = 0.f;
    for (int c = 0; c < C0; ++c) acc += srow[c] * W[c * 32 + d];
    float* O = (mat == 0) ? qb : ((mat == 1) ? kb : ((mat == 2) ? v0b : ssb));
    O[l * 32 + d] = acc;
  }
  for (int t = tid; t < 192; t += 128) {
    int sel = t / 96, rem = t % 96;
    int d = rem / 3, x = rem % 3;
    const float* W = sel ? wself1 : wv1;
    float acc = 0.f;
    for (int c = 0; c < C1; ++c) acc += vrow[c * 3 + x] * W[c * 32 + d];
    (sel ? vs : vm)[l * 96 + d * 3 + x] = acc;
  }
}

// ---------------------------------------------------------------------------
// K3b: attention layer. One block per query row i. TDM DMA of the 512x64 f16
// edge row-block into LDS (overlapped with weight staging), WMMA projections
// read A fragments directly from the linear LDS block, full-row softmax,
// fused s/v updates + LN.
// ---------------------------------------------------------------------------
__global__ __launch_bounds__(256) void attn_kernel(
    const f16* __restrict__ edge, const float* __restrict__ qb,
    const float* __restrict__ kb, const float* __restrict__ v0b,
    const float* __restrict__ sself, const float* __restrict__ vm,
    const float* __restrict__ vself, const float* __restrict__ xyz,
    const float* __restrict__ wek, const float* __restrict__ wev0,
    const float* __restrict__ wr1, const float* __restrict__ wr2,
    const float* __restrict__ lng, const float* __restrict__ lnb,
    float* __restrict__ s_out, float* __restrict__ v_out) {
  extern __shared__ __align__(32) char smem[];
  char* sp = smem;
  auto carve = [&](size_t n) {
    char* p = sp;
    sp += (n + 31) & ~size_t(31);
    return p;
  };
  f16* eblk = (f16*)carve(512 * 64 * sizeof(f16));    // full edge row-block
  f16* wst = (f16*)carve(4 * 4 * 512 * sizeof(f16));  // 4 mats, B layout
  float* att = (float*)carve(512 * 4 * 4);            // logits -> probs
  float* am = (float*)carve(512 * 4);
  float* eks = (float*)carve(32 * 32 * 4);
  float* evs = (float*)carve(32 * 32 * 4);
  float* r1s = (float*)carve(32 * 32 * 4);
  float* r2s = (float*)carve(32 * 32 * 4);
  float* unl = (float*)carve(32 * 3 * 4);
  float* sq = (float*)carve(32 * 4);
  float* mh = (float*)carve(4 * 4);
  float* zh = (float*)carve(4 * 4);

  const int tid = threadIdx.x, wv = tid >> 5, lane = tid & 31;
  const int i = blockIdx.x;
  const f16* erow = edge + ((size_t)i * 512) * 64;

#ifdef HAVE_TDM
  // Issue the DMA first so it overlaps the weight staging below.
  if (wv == 0) {
    unsigned lds_off = (unsigned)((const char*)eblk - smem);
    unsigned long long ga = (unsigned long long)(uintptr_t)erow;
    u32x4 g0;
    g0[0] = 1u;                              // count=1, user descriptor
    g0[1] = lds_off;                         // lds_addr (bytes)
    g0[2] = (unsigned)ga;                    // global_addr[31:0]
    g0[3] = (unsigned)((ga >> 32) & 0x1FFFFFFu) | (2u << 30);  // addr hi|type=2
    i32x8 g1;
    g1[0] = (1 << 16);                       // data_size = 2 bytes
    g1[1] = (int)(64u << 16);                // tensor_dim0 = 64   (bits 79:48)
    g1[2] = (int)(512u << 16);               // tensor_dim1 = 512  (bits 111:80)
    g1[3] = (int)(64u << 16);                // tile_dim0 = 64     (bits 127:112)
    g1[4] = 512;                             // tile_dim1 = 512, tile_dim2 = 0
    g1[5] = 64;                              // tensor_dim0_stride = 64
    g1[6] = 0;
    g1[7] = 0;
    i32x4 g2 = {0, 0, 0, 0};
    i32x4 g3 = {0, 0, 0, 0};
    i32x8 g4 = {0, 0, 0, 0, 0, 0, 0, 0};     // extra group (clang-23 6-arg form)
    __builtin_amdgcn_tensor_load_to_lds(g0, g1, g2, g3, g4, 0);
  }
#else
  for (int t = tid; t < 4096; t += 256)
    *(u32x4*)(eblk + t * 8) = *(const u32x4*)(erow + t * 8);
#endif

  const float* Ws[4] = {wek, wev0, wr1, wr2};
  for (int t = tid; t < 4 * 64 * 32; t += 256) {
    int mat = t >> 11, rem = t & 2047, k = rem >> 5, n = rem & 31;
    wst[(mat * 4 + ((k >> 5) << 1) + (n >> 4)) * 512 + b_slot(k & 31, n & 15)] =
        (f16)Ws[mat][k * 32 + n];
  }
  if (tid < 32) sq[tid] = qb[i * 32 + tid];
#ifdef HAVE_TDM
  if (wv == 0) __builtin_amdgcn_s_wait_tensorcnt(0);
#endif
  __syncthreads();

  const int m16 = lane & 15;
  const int hi8 = (lane >> 4) << 3;

  // ---- Pass A: logits (32 j per iteration; 4 waves do WMMA) ----
  for (int jt = 0; jt < 16; ++jt) {
    const int j0 = jt * 32;
    if (wv < 4) {
      const int tile = wv >> 1, nt = wv & 1;
      const f16* row = eblk + (j0 + tile * 16 + m16) * 64;
      v8f acc = {};
#pragma unroll
      for (int c = 0; c < 2; ++c) {
        v16h a = load_a_frag(row, 32 * c + hi8);
        v16h b = *(const v16h*)(wst + ((c << 1) + nt) * 512 + lane * 16);
        acc = wmma16(a, b, acc);
      }
      const int n0 = nt * 16 + (lane & 15);
      const int mb = tile * 16 + ((lane >> 4) << 3);
#pragma unroll
      for (int r = 0; r < 8; ++r) {
        int jl = mb + r;
        eks[jl * 32 + n0] = acc[r] + kb[(j0 + jl) * 32 + n0];
      }
    }
    __syncthreads();
    if (tid < 128) {
      int jl = tid >> 2, h = tid & 3;
      float l = 0.f;
      for (int d = 0; d < 8; ++d) l += sq[h * 8 + d] * eks[jl * 32 + h * 8 + d];
      l *= 0.35355339059327376f;  // 1/sqrt(DH)
      if (j0 + jl == i) l = -1e9f;
      att[(j0 + jl) * 4 + h] = l;
    }
    __syncthreads();
  }

  // ---- Softmax over j, per head (one wave per head) ----
  if (wv < 4) {
    int h = wv;
    float mx = -3.4e38f;
    for (int j = lane; j < 512; j += 32) mx = fmaxf(mx, att[j * 4 + h]);
    mx = wmax32(mx);
    float z = 0.f;
    for (int j = lane; j < 512; j += 32) z += __expf(att[j * 4 + h] - mx);
    z = wsum32(z);
    if (lane == 0) { mh[h] = mx; zh[h] = z; }
  }
  __syncthreads();
  for (int t = tid; t < 2048; t += 256) {
    int h = t & 3;
    att[t] = __expf(att[t] - mh[h]) / zh[h];
  }
  __syncthreads();
  for (int j = tid; j < 512; j += 256)
    am[j] =
        0.25f * (att[j * 4] + att[j * 4 + 1] + att[j * 4 + 2] + att[j * 4 + 3]);
  __syncthreads();

  const float cix = xyz[i * 9 + 3], ciy = xyz[i * 9 + 4], ciz = xyz[i * 9 + 5];
  float sacc = 0.f, vacc = 0.f;

  // ---- Pass B: value / vector updates (32 j per iteration) ----
  for (int jt = 0; jt < 16; ++jt) {
    const int j0 = jt * 32;
    if (wv < 6) {
      for (int t = wv; t < 12; t += 6) {
        const int tile = (t >= 6) ? 1 : 0;
        const int q = t % 6;
        const int mloc = q >> 1;  // 0=ev0, 1=r1, 2=r2
        const int nt = q & 1;
        float* outp = (mloc == 0) ? evs : ((mloc == 1) ? r1s : r2s);
        const f16* row = eblk + (j0 + tile * 16 + m16) * 64;
        v8f acc = {};
#pragma unroll
        for (int c = 0; c < 2; ++c) {
          v16h a = load_a_frag(row, 32 * c + hi8);
          v16h b = *(const v16h*)(wst + ((mloc + 1) * 4 + (c << 1) + nt) * 512 +
                                  lane * 16);
          acc = wmma16(a, b, acc);
        }
        const int n0 = nt * 16 + (lane & 15);
        const int mb = tile * 16 + ((lane >> 4) << 3);
#pragma unroll
        for (int r = 0; r < 8; ++r) {
          int jl = mb + r;
          float v = acc[r];
          if (mloc == 0) v += v0b[(j0 + jl) * 32 + n0];
          outp[jl * 32 + n0] = v;
        }
      }
    } else if (wv == 6) {
      int j = j0 + lane;
      float dx = xyz[j * 9 + 3] - cix, dy = xyz[j * 9 + 4] - ciy,
            dz = xyz[j * 9 + 5] - ciz;
      float D = sqrtf(dx * dx + dy * dy + dz * dz + 1e-8f);
      float inv = 1.f / D;
      unl[lane * 3 + 0] = dx * inv;
      unl[lane * 3 + 1] = dy * inv;
      unl[lane * 3 + 2] = dz * inv;
    }
    __syncthreads();
    if (tid < 32) {
      int h = tid >> 3;
      for (int jl = 0; jl < 32; ++jl)
        sacc += att[(j0 + jl) * 4 + h] * evs[jl * 32 + tid];
    } else if (tid < 128) {
      int c = (tid - 32) & 31, x = (tid - 32) >> 5;
      for (int jl = 0; jl < 32; ++jl) {
        int j = j0 + jl;
        vacc += am[j] * (r1s[jl * 32 + c] * vm[j * 96 + c * 3 + x] +
                         r2s[jl * 32 + c] * unl[jl * 3 + x]);
      }
    }
    __syncthreads();
  }

  if (tid < 32) {
    float sv = sacc + sself[i * 32 + tid];
    float mean = wsum32(sv) * (1.f / 32.f);
    float d = sv - mean;
    float var = wsum32(d * d) * (1.f / 32.f);
    s_out[i * 32 + tid] = d * rsqrtf(var + 1e-5f) * lng[tid] + lnb[tid];
  } else if (tid < 128) {
    int c = (tid - 32) & 31, x = (tid - 32) >> 5;
    v_out[i * 96 + c * 3 + x] = vacc + vself[i * 96 + c * 3 + x];
  }
}

// ---------------------------------------------------------------------------
// K4: outputs: state_out = state + s@wout0 ; quaternion frame update for xyz.
// ---------------------------------------------------------------------------
__global__ __launch_bounds__(64) void out_kernel(
    const float* __restrict__ s, const float* __restrict__ v,
    const float* __restrict__ state, const float* __restrict__ xyz,
    const float* __restrict__ wout0, const float* __restrict__ wout1,
    float* __restrict__ out_state, float* __restrict__ out_xyz) {
  __shared__ float srow[32];
  __shared__ float vrow[96];
  __shared__ float s1[6];
  const int l = blockIdx.x, tid = threadIdx.x;
  if (tid < 32) srow[tid] = s[l * 32 + tid];
  for (int t = tid; t < 96; t += 64) vrow[t] = v[l * 96 + t];
  __syncthreads();
  {
    float acc = 0.f;
    for (int c = 0; c < 32; ++c) acc += srow[c] * wout0[c * 64 + tid];
    out_state[l * 64 + tid] = state[l * 64 + tid] + acc;
  }
  if (tid < 6) {
    int o = tid / 3, x = tid % 3;
    float acc = 0.f;
    for (int c = 0; c < 32; ++c) acc += vrow[c * 3 + x] * wout1[c * 2 + o];
    s1[o * 3 + x] = acc;
  }
  __syncthreads();
  if (tid < 9) {
    float T[3] = {s1[0] * 0.1f, s1[1] * 0.1f, s1[2] * 0.1f};
    float Rx = s1[3] * 0.01f, Ry = s1[4] * 0.01f, Rz = s1[5] * 0.01f;
    float Qn = sqrtf(1.f + Rx * Rx + Ry * Ry + Rz * Rz);
    float qA = 1.f / Qn, qB = Rx / Qn, qC = Ry / Qn, qD = Rz / Qn;
    float R[3][3] = {
        {qA * qA + qB * qB - qC * qC - qD * qD, 2 * qB * qC - 2 * qA * qD,
         2 * qB * qD + 2 * qA * qC},
        {2 * qB * qC + 2 * qA * qD, qA * qA - qB * qB + qC * qC - qD * qD,
         2 * qC * qD - 2 * qA * qB},
        {2 * qB * qD - 2 * qA * qC, 2 * qC * qD + 2 * qA * qB,
         qA * qA - qB * qB - qC * qC + qD * qD}};
    int a = tid / 3, ic = tid % 3;
    float acc = 0.f;
    for (int jj = 0; jj < 3; ++jj) {
      float vc = xyz[l * 9 + a * 3 + jj] - xyz[l * 9 + 3 + jj];
      acc += R[ic][jj] * vc;
    }
    out_xyz[l * 9 + a * 3 + ic] = acc + xyz[l * 9 + 3 + ic] + T[ic];
  }
}

// ---------------------------------------------------------------------------
// Host launcher. Input order = setup_inputs() dict insertion order:
// msa, pair, state, xyz, <params in insertion order>, idx.
// ---------------------------------------------------------------------------
extern "C" void kernel_launch(void* const* d_in, const int* in_sizes, int n_in,
                              void* d_out, int out_size, void* d_ws,
                              size_t ws_size, hipStream_t stream) {
  (void)in_sizes; (void)n_in; (void)out_size; (void)ws_size;
  const float* msa = (const float*)d_in[0];
  const float* pair = (const float*)d_in[1];
  const float* state = (const float*)d_in[2];
  const float* xyz = (const float*)d_in[3];
  const float* nmg = (const float*)d_in[4];   // norm_msa_g
  const float* nmb = (const float*)d_in[5];
  const float* nsg = (const float*)d_in[6];   // norm_state_g
  const float* nsb = (const float*)d_in[7];
  const float* nng = (const float*)d_in[8];   // norm_node_g
  const float* nnb = (const float*)d_in[9];
  const float* npg = (const float*)d_in[10];  // norm_pair_g
  const float* npb = (const float*)d_in[11];
  const float* neg = (const float*)d_in[12];  // norm_edge_g
  const float* neb = (const float*)d_in[13];
  const float* enw = (const float*)d_in[14];  // embed_node_w
  const float* enb = (const float*)d_in[15];
  const float* eew = (const float*)d_in[16];  // embed_edge_w
  const float* eeb = (const float*)d_in[17];
  const float* fn_ng = (const float*)d_in[18];
  const float* fn_nb = (const float*)d_in[19];
  const float* fn_w1 = (const float*)d_in[20];
  const float* fn_b1 = (const float*)d_in[21];
  const float* fn_w2 = (const float*)d_in[22];
  const float* fn_b2 = (const float*)d_in[23];
  const float* fe_ng = (const float*)d_in[24];
  const float* fe_nb = (const float*)d_in[25];
  const float* fe_w1 = (const float*)d_in[26];
  const float* fe_b1 = (const float*)d_in[27];
  const float* fe_w2 = (const float*)d_in[28];
  const float* fe_b2 = (const float*)d_in[29];
  // l0 / l1 blocks: wq wk wek wv0 wev0 wv1 wr1 wr2 wself0 wself1 ng nb
  const float* L0[12];
  const float* L1[12];
  for (int t = 0; t < 12; ++t) {
    L0[t] = (const float*)d_in[30 + t];
    L1[t] = (const float*)d_in[42 + t];
  }
  const float* wout0 = (const float*)d_in[54];
  const float* wout1 = (const float*)d_in[55];
  const int* idx = (const int*)d_in[56];

  // Workspace: edge f16 (32MB) then f32 region.
  f16* edge = (f16*)d_ws;
  float* base = (float*)((char*)d_ws + (size_t)LSEQ * LSEQ * 64 * sizeof(f16));
  float* node = base;           // 512*64
  float* sA = node + 512 * 64;  // 512*32
  float* sB = sA + 512 * 32;    // 512*32
  float* vA = sB + 512 * 32;    // 512*96
  float* vB = vA + 512 * 96;    // 512*96
  float* qb = vB + 512 * 96;    // 512*32
  float* kb = qb + 512 * 32;
  float* v0b = kb + 512 * 32;
  float* ssb = v0b + 512 * 32;
  float* vm = ssb + 512 * 32;  // 512*96
  float* vs = vm + 512 * 96;   // 512*96

  node_kernel<<<512, 64, 0, stream>>>(msa, state, nmg, nmb, nsg, nsb, enw, enb,
                                      fn_ng, fn_nb, fn_w1, fn_b1, fn_w2, fn_b2,
                                      nng, nnb, node);

  edge_kernel<<<(512 * 512) / 32, 256, 90112, stream>>>(
      pair, xyz, idx, npg, npb, eew, eeb, fe_ng, fe_nb, fe_w1, fe_b1, fe_w2,
      fe_b2, neg, neb, edge);

  // layer 0: s dim 64 (node), v from xyz (C1=3)
  prep_kernel<<<512, 128, 0, stream>>>(node, 64, nullptr, 3, xyz, L0[0], L0[1],
                                       L0[3], L0[8], L0[5], L0[9], qb, kb, v0b,
                                       ssb, vm, vs);
  attn_kernel<<<512, 256, 110592, stream>>>(edge, qb, kb, v0b, ssb, vm, vs, xyz,
                                            L0[2], L0[4], L0[6], L0[7], L0[10],
                                            L0[11], sA, vA);
  // layer 1: s dim 32, v dim 32
  prep_kernel<<<512, 128, 0, stream>>>(sA, 32, vA, 32, xyz, L1[0], L1[1], L1[3],
                                       L1[8], L1[5], L1[9], qb, kb, v0b, ssb,
                                       vm, vs);
  attn_kernel<<<512, 256, 110592, stream>>>(edge, qb, kb, v0b, ssb, vm, vs, xyz,
                                            L1[2], L1[4], L1[6], L1[7], L1[10],
                                            L1[11], sB, vB);

  out_kernel<<<512, 64, 0, stream>>>(sB, vB, state, xyz, wout0, wout1,
                                     (float*)d_out, (float*)d_out + 512 * 64);
}
